// GatedDeltaNet_13872744366525
// MI455X (gfx1250) — matrix operand order repaired and checked by reading, not compiled, not run on hardware
//
#include <hip/hip_runtime.h>

// ---------------- types ----------------
typedef __attribute__((ext_vector_type(16))) __bf16 v16bf;
typedef __attribute__((ext_vector_type(8)))  __bf16 v8bf;
typedef __attribute__((ext_vector_type(8)))  float  v8f;

// ---------------- problem constants ----------------
#define BB   2
#define SS   2048
#define HH   2048
#define NKH  8
#define NVH  16
#define DKD  64
#define DVD  64
#define KSZ  4
#define KDIM 512          // NK*DK
#define VDIM 1024         // NV*DV
#define CDIM 2048         // 2*KDIM + VDIM
#define MM   (BB*SS)      // 4096 rows
#define LDC1 3104         // CDIM + VDIM + NV + NV

// =====================================================================
// f32 -> bf16 cast
// =====================================================================
__global__ __launch_bounds__(256) void cast_bf16_kernel(const float* __restrict__ in,
                                                        __bf16* __restrict__ out, long n) {
  long i = (long)blockIdx.x * 256 + threadIdx.x;
  if (i < n) out[i] = (__bf16)in[i];
}

// =====================================================================
// async copy of 16 bytes/lane from global to LDS (ASYNCcnt-tracked DMA)
// =====================================================================
__device__ __forceinline__ void async_copy_b128(const __bf16* gsrc, __bf16* ldst) {
  unsigned           lds_addr = (unsigned)(unsigned long long)(void*)ldst;
  unsigned long long gaddr    = (unsigned long long)(const void*)gsrc;
  asm volatile("global_load_async_to_lds_b128 %0, %1, off"
               :: "v"(lds_addr), "v"(gaddr) : "memory");
}

// =====================================================================
// bf16 WMMA GEMM:  C[M x ldc slice] = A[M x K] * W[N x K]^T  (f32 accum)
// grid.x = N/32, grid.y = M/256, block = 256 (8 waves, wave32)
// wave computes a 32x32 tile; B tiles (32n x 64k) double-buffered in LDS
// via global_load_async_to_lds_b128 (ASYNCcnt), 8 WMMAs per stage.
// =====================================================================
__global__ __launch_bounds__(256) void gemm_bf16_wmma_kernel(
    const __bf16* __restrict__ A, const __bf16* __restrict__ W,
    float* __restrict__ C, int K, int ldc) {
  __shared__ __align__(32) __bf16 lds_b[2][32 * 64];   // [buf][n 0..31][k 0..63]

  const int tid  = threadIdx.x;
  const int wave = tid >> 5;
  const int lane = tid & 31;
  const int lr   = lane & 15;      // row/col within 16
  const int lh   = lane >> 4;      // half select

  const int m0 = blockIdx.y * 256 + wave * 32;
  const int n0 = blockIdx.x * 32;

  v8f acc00 = {}; v8f acc01 = {}; v8f acc10 = {}; v8f acc11 = {};

  const int srow = tid >> 3;           // 0..31  (n within tile)
  const int skk  = (tid & 7) * 8;      // 0..56  (k within tile, 8 bf16 = 16B)
  const __bf16* wsrc = W + (size_t)(n0 + srow) * K + skk;

  const int nk = K >> 6;               // number of 64-wide K tiles

  // prologue: kick off tile 0 DMA
  async_copy_b128(wsrc, &lds_b[0][srow * 64 + skk]);

  for (int it = 0; it < nk; ++it) {
    __syncthreads();   // readers of the buffer we are about to overwrite are done
    if (it + 1 < nk) {
      async_copy_b128(wsrc + (it + 1) * 64, &lds_b[(it + 1) & 1][srow * 64 + skk]);
      asm volatile("s_wait_asynccnt 0x1" ::: "memory");   // tile `it` landed
    } else {
      asm volatile("s_wait_asynccnt 0x0" ::: "memory");
    }
    __syncthreads();   // tile `it` visible to all waves

    const __bf16* lb = lds_b[it & 1];
    const int k0 = it * 64;
    #pragma unroll
    for (int ks = 0; ks < 64; ks += 32) {
      // A fragments (lane = m, 16 contiguous K at ks + 16*half): two 16B chunks per lane
      const __bf16* arow0 = A + (size_t)(m0 + lr) * K + k0 + ks;
      const __bf16* arow1 = A + (size_t)(m0 + 16 + lr) * K + k0 + ks;
      v8bf a0lo = *(const v8bf*)(arow0 + 16 * lh);
      v8bf a0hi = *(const v8bf*)(arow0 + 16 * lh + 8);
      v8bf a1lo = *(const v8bf*)(arow1 + 16 * lh);
      v8bf a1hi = *(const v8bf*)(arow1 + 16 * lh + 8);
      v16bf a0 = __builtin_shufflevector(a0lo, a0hi, 0,1,2,3,4,5,6,7,8,9,10,11,12,13,14,15);
      v16bf a1 = __builtin_shufflevector(a1lo, a1hi, 0,1,2,3,4,5,6,7,8,9,10,11,12,13,14,15);

      // B fragments (lane = n, 16 contiguous K at ks + 16*half) from LDS
      v16bf b0 = *(const v16bf*)&lb[lr * 64 + ks + 16 * lh];
      v16bf b1 = *(const v16bf*)&lb[(16 + lr) * 64 + ks + 16 * lh];

      acc00 = __builtin_amdgcn_wmma_f32_16x16x32_bf16(false, a0, false, b0, (short)0, acc00, false, false);
      acc01 = __builtin_amdgcn_wmma_f32_16x16x32_bf16(false, a0, false, b1, (short)0, acc01, false, false);
      acc10 = __builtin_amdgcn_wmma_f32_16x16x32_bf16(false, a1, false, b0, (short)0, acc10, false, false);
      acc11 = __builtin_amdgcn_wmma_f32_16x16x32_bf16(false, a1, false, b1, (short)0, acc11, false, false);
    }
  }

  // store: D layout -> lane holds col n=lr (half lh picks M base 0/8), vgpr r = M%8
  #pragma unroll
  for (int r = 0; r < 8; r++) {
    size_t mrow = (size_t)(m0 + 8 * lh + r);
    C[mrow * ldc + n0 + lr]              = acc00[r];
    C[mrow * ldc + n0 + 16 + lr]         = acc01[r];
    C[(mrow + 16) * ldc + n0 + lr]       = acc10[r];
    C[(mrow + 16) * ldc + n0 + 16 + lr]  = acc11[r];
  }
}

// =====================================================================
// depthwise causal conv (K=4) + SiLU over the qkv slice of C1
// =====================================================================
__global__ __launch_bounds__(256) void conv_silu_kernel(const float* __restrict__ c1,
                                                        const float* __restrict__ w,
                                                        float* __restrict__ qc) {
  long idx = (long)blockIdx.x * 256 + threadIdx.x;  // over MM*CDIM
  int  c   = (int)(idx & (CDIM - 1));
  long row = idx >> 11;
  int  s   = (int)(row & (SS - 1));
  float acc = 0.f;
  #pragma unroll
  for (int j = 0; j < KSZ; j++) {
    int sp = s - (KSZ - 1) + j;
    if (sp >= 0) acc += w[c * KSZ + j] * c1[(row - (KSZ - 1) + j) * LDC1 + c];
  }
  acc = acc / (1.f + __expf(-acc));     // SiLU
  qc[idx] = acc;
}

// =====================================================================
// beta = sigmoid(b_raw), g = -exp(A_log)*softplus(a_raw + dt_bias)
// =====================================================================
__global__ __launch_bounds__(256) void bg_kernel(const float* __restrict__ c1,
                                                 const float* __restrict__ dtb,
                                                 const float* __restrict__ alog,
                                                 float* __restrict__ gb,
                                                 float* __restrict__ bb) {
  int i = blockIdx.x * 256 + threadIdx.x;   // MM*NV
  if (i >= MM * NVH) return;
  int row = i >> 4, h = i & 15;
  float braw = c1[(long)row * LDC1 + CDIM + VDIM + h];
  bb[i] = 1.f / (1.f + __expf(-braw));
  float ar = c1[(long)row * LDC1 + CDIM + VDIM + NVH + h] + dtb[h];
  float sp = (ar > 20.f) ? ar : log1pf(__expf(ar));
  gb[i] = -__expf(alog[h]) * sp;
}

// =====================================================================
// L2-normalize q and k heads (wave per head-row, 2 elems/lane)
// =====================================================================
__global__ __launch_bounds__(256) void l2norm_kernel(const float* __restrict__ qc,
                                                     float* __restrict__ qn,
                                                     float* __restrict__ kn) {
  int gid  = blockIdx.x * 8 + (threadIdx.x >> 5);   // MM*NK*2 waves
  int lane = threadIdx.x & 31;
  int which = gid & 1;          // 0=q, 1=k
  int rh    = gid >> 1;
  int row   = rh >> 3, hd = rh & 7;
  long src  = (long)row * CDIM + which * KDIM + hd * DKD + lane * 2;
  float x0 = qc[src], x1 = qc[src + 1];
  float ss = x0 * x0 + x1 * x1;
  #pragma unroll
  for (int off = 16; off; off >>= 1) ss += __shfl_xor(ss, off, 32);
  float r = rsqrtf(ss + 1e-6f);
  if (which == 0) r *= 0.125f;  // DK^-0.5
  float* dst = which ? kn : qn;
  long db = (long)row * KDIM + hd * DKD + lane * 2;
  dst[db]     = x0 * r;
  dst[db + 1] = x1 * r;
}

// =====================================================================
// gated delta-rule scan: one block per (b, head_v); 256 threads;
// thread owns state rows kg*16..kg*16+15 of column v  (v=tid>>2, kg=tid&3
// so the 4 K-partials reduce inside a wave with shfl_xor)
// =====================================================================
__global__ __launch_bounds__(256) void scan_kernel(const float* __restrict__ qn,
                                                   const float* __restrict__ kn,
                                                   const float* __restrict__ qc,
                                                   const float* __restrict__ gb,
                                                   const float* __restrict__ bb,
                                                   float* __restrict__ o) {
  const int head = blockIdx.x & 15;
  const int b    = blockIdx.x >> 4;
  const int tid  = threadIdx.x;
  const int v    = tid >> 2;        // 0..63
  const int kg   = tid & 3;         // 0..3
  const int hk   = head >> 1;       // repeat_interleave(rep=2)

  __shared__ float qL[DKD], kL[DKD], vL[DVD], sc[2];

  float s[16];
  #pragma unroll
  for (int i = 0; i < 16; i++) s[i] = 0.f;

  for (int t = 0; t < SS; t++) {
    long row = (long)b * SS + t;
    if (tid < 64)        qL[tid]       = qn[row * KDIM + hk * DKD + tid];
    else if (tid < 128)  kL[tid - 64]  = kn[row * KDIM + hk * DKD + (tid - 64)];
    else if (tid < 192)  vL[tid - 128] = qc[row * CDIM + 2 * KDIM + head * DVD + (tid - 128)];
    else if (tid == 192) { sc[0] = __expf(gb[row * NVH + head]); sc[1] = bb[row * NVH + head]; }
    __syncthreads();

    const float eg = sc[0], bt = sc[1];
    float kv = 0.f;
    #pragma unroll
    for (int i = 0; i < 16; i++) { s[i] *= eg; kv += s[i] * kL[kg * 16 + i]; }
    kv += __shfl_xor(kv, 1, 32);
    kv += __shfl_xor(kv, 2, 32);
    const float d = (vL[v] - kv) * bt;

    float ov = 0.f;
    #pragma unroll
    for (int i = 0; i < 16; i++) { s[i] += kL[kg * 16 + i] * d; ov += s[i] * qL[kg * 16 + i]; }
    ov += __shfl_xor(ov, 1, 32);
    ov += __shfl_xor(ov, 2, 32);
    if (kg == 0) o[row * VDIM + head * DVD + v] = ov;
    __syncthreads();
  }
}

// =====================================================================
// RMSNorm over DV + SiLU(z) gate, emit bf16 for the output GEMM
// =====================================================================
__global__ __launch_bounds__(256) void norm_gate_kernel(const float* __restrict__ o,
                                                        const float* __restrict__ c1,
                                                        const float* __restrict__ nw,
                                                        __bf16* __restrict__ og) {
  int gid  = blockIdx.x * 8 + (threadIdx.x >> 5);   // MM*NV waves
  int lane = threadIdx.x & 31;
  int row  = gid >> 4, h = gid & 15;
  long ob  = (long)row * VDIM + h * DVD + lane * 2;
  float o0 = o[ob], o1 = o[ob + 1];
  float ss = o0 * o0 + o1 * o1;
  #pragma unroll
  for (int off = 16; off; off >>= 1) ss += __shfl_xor(ss, off, 32);
  float r = rsqrtf(ss * (1.f / DVD) + 1e-6f);
  float w0 = 1.f + nw[lane * 2], w1 = 1.f + nw[lane * 2 + 1];
  long zb = (long)row * LDC1 + CDIM + h * DVD + lane * 2;
  float z0 = c1[zb], z1 = c1[zb + 1];
  float g0 = z0 / (1.f + __expf(-z0));
  float g1 = z1 / (1.f + __expf(-z1));
  og[ob]     = (__bf16)(o0 * r * w0 * g0);
  og[ob + 1] = (__bf16)(o1 * r * w1 * g1);
}

// =====================================================================
extern "C" void kernel_launch(void* const* d_in, const int* in_sizes, int n_in,
                              void* d_out, int out_size, void* d_ws, size_t ws_size,
                              hipStream_t stream) {
  const float* hs   = (const float*)d_in[0];
  const float* Wqkv = (const float*)d_in[1];
  const float* convw= (const float*)d_in[2];
  const float* Wz   = (const float*)d_in[3];
  const float* Wb   = (const float*)d_in[4];
  const float* Wa   = (const float*)d_in[5];
  const float* Wout = (const float*)d_in[6];
  const float* dtb  = (const float*)d_in[7];
  const float* alog = (const float*)d_in[8];
  const float* nw   = (const float*)d_in[9];
  float* out = (float*)d_out;

  char* ws = (char*)d_ws;
  size_t off = 0;
  auto carve = [&](size_t bytes) { size_t o = off; off += (bytes + 255) & ~(size_t)255; return o; };

  __bf16* hsb   = (__bf16*)(ws + carve((size_t)MM * HH * 2));
  __bf16* wcat  = (__bf16*)(ws + carve((size_t)LDC1 * HH * 2));
  __bf16* woutb = (__bf16*)(ws + carve((size_t)HH * VDIM * 2));
  float*  c1    = (float*)(ws + carve((size_t)MM * LDC1 * 4));
  float*  qc    = (float*)(ws + carve((size_t)MM * CDIM * 4));
  float*  qn    = (float*)(ws + carve((size_t)MM * KDIM * 4));
  float*  kn    = (float*)(ws + carve((size_t)MM * KDIM * 4));
  float*  gbuf  = (float*)(ws + carve((size_t)MM * NVH * 4));
  float*  bbuf  = (float*)(ws + carve((size_t)MM * NVH * 4));
  float*  obuf  = (float*)(ws + carve((size_t)MM * VDIM * 4));
  __bf16* ogb   = (__bf16*)(ws + carve((size_t)MM * VDIM * 2));

  // ---- 1) cast activations + concatenated weights to bf16 ----
  cast_bf16_kernel<<<(MM * HH) / 256, 256, 0, stream>>>(hs, hsb, (long)MM * HH);
  cast_bf16_kernel<<<(CDIM * HH) / 256, 256, 0, stream>>>(Wqkv, wcat, (long)CDIM * HH);
  cast_bf16_kernel<<<(VDIM * HH) / 256, 256, 0, stream>>>(Wz, wcat + (size_t)CDIM * HH, (long)VDIM * HH);
  cast_bf16_kernel<<<(NVH * HH) / 256, 256, 0, stream>>>(Wb, wcat + (size_t)(CDIM + VDIM) * HH, (long)NVH * HH);
  cast_bf16_kernel<<<(NVH * HH) / 256, 256, 0, stream>>>(Wa, wcat + (size_t)(CDIM + VDIM + NVH) * HH, (long)NVH * HH);
  cast_bf16_kernel<<<(HH * VDIM) / 256, 256, 0, stream>>>(Wout, woutb, (long)HH * VDIM);

  // ---- 2) fused projections: C1[4096 x 3104] = hs * [Wqkv;Wz;Wb;Wa]^T ----
  gemm_bf16_wmma_kernel<<<dim3(LDC1 / 32, MM / 256), 256, 0, stream>>>(hsb, wcat, c1, HH, LDC1);

  // ---- 3) depthwise causal conv + SiLU ----
  conv_silu_kernel<<<(MM * CDIM) / 256, 256, 0, stream>>>(c1, convw, qc);

  // ---- 4) beta / g gates ----
  bg_kernel<<<(MM * NVH) / 256, 256, 0, stream>>>(c1, dtb, alog, gbuf, bbuf);

  // ---- 5) l2-normalize q,k ----
  l2norm_kernel<<<(MM * NKH * 2) / 8, 256, 0, stream>>>(qc, qn, kn);

  // ---- 6) sequential gated delta-rule scan ----
  scan_kernel<<<BB * NVH, 256, 0, stream>>>(qn, kn, qc, gbuf, bbuf, obuf);

  // ---- 7) RMSNorm + SiLU(z) gate -> bf16 ----
  norm_gate_kernel<<<(MM * NVH) / 8, 256, 0, stream>>>(obuf, c1, nw, ogb);

  // ---- 8) output projection: out[4096 x 2048] = og * Wout^T ----
  gemm_bf16_wmma_kernel<<<dim3(HH / 32, MM / 256), 256, 0, stream>>>(ogb, woutb, out, VDIM, HH);
}
